// CrossGraphFusionLayer_17300128268551
// MI455X (gfx1250) — matrix operand order repaired and checked by reading, not compiled, not run on hardware
//
#include <hip/hip_runtime.h>
#include <math.h>

#define BB 48   // image batch
#define NC 48   // number of captions (cap dim 0)
#define RR 36   // image regions
#define LL 24   // caption length
#define DD 1024 // feature dim
#define CDIV(a,b) (((a)+(b)-1)/(b))

typedef __attribute__((ext_vector_type(16))) __bf16 v16bf;
typedef __attribute__((ext_vector_type(8)))  __bf16 v8bf;
typedef __attribute__((ext_vector_type(8)))  float  v8f;

// ---------- bf16 helpers ----------
__device__ __forceinline__ unsigned short f2bf(float f) {
  unsigned int u = __float_as_uint(f);
  unsigned int r = u + 0x7FFFu + ((u >> 16) & 1u);
  return (unsigned short)(r >> 16);
}

// A fragment (16x32, 16-bit): lane = m + 16*h holds K = {8h..8h+7} U {16+8h..23+8h}
__device__ __forceinline__ v16bf loadFragA(const unsigned short* row, int k0, int h) {
  v8bf a0 = *(const v8bf*)(row + k0 + 8 * h);
  v8bf a1 = *(const v8bf*)(row + k0 + 16 + 8 * h);
  return __builtin_shufflevector(a0, a1, 0,1,2,3,4,5,6,7,8,9,10,11,12,13,14,15);
}
// B fragment (32x16, 16-bit, K-major source row = column n of B): lane = n + 16*h holds K = {8h..8h+7} U {16+8h..23+8h}
__device__ __forceinline__ v16bf loadFragB(const unsigned short* row, int k0, int h) {
  v8bf b0 = *(const v8bf*)(row + k0 + 8 * h);
  v8bf b1 = *(const v8bf*)(row + k0 + 16 + 8 * h);
  return __builtin_shufflevector(b0, b1, 0,1,2,3,4,5,6,7,8,9,10,11,12,13,14,15);
}

// ---------- cast ----------
__global__ void cast_bf16_kernel(const float* __restrict__ src, unsigned short* __restrict__ dst, int n) {
  int i = blockIdx.x * blockDim.x + threadIdx.x;
  int stride = gridDim.x * blockDim.x;
  for (; i < n; i += stride) dst[i] = f2bf(src[i]);
}

// ---------- linear: Out(M,N) = A(M,K) @ W(N,K)^T + bias, bf16 out; M%32==0, N%64==0, K%32==0 ----------
__global__ void linear_wmma_kernel(const unsigned short* __restrict__ A,
                                   const unsigned short* __restrict__ W,
                                   const float* __restrict__ bias,
                                   unsigned short* __restrict__ Out,
                                   int M, int N, int K) {
  int lane = threadIdx.x & 31;
  int wv = threadIdx.x >> 5;
  int h = lane >> 4, ln = lane & 15;
  int m0 = blockIdx.x * 32 + (wv >> 2) * 16;
  int n0 = blockIdx.y * 64 + (wv & 3) * 16;
  const unsigned short* arow = A + (size_t)(m0 + ln) * K;
  const unsigned short* wrow = W + (size_t)(n0 + ln) * K;
  v8f acc = {0.f, 0.f, 0.f, 0.f, 0.f, 0.f, 0.f, 0.f};
  for (int k0 = 0; k0 < K; k0 += 32) {
    __builtin_prefetch(arow + k0 + 512, 0, 1);
    v16bf a = loadFragA(arow, k0, h);
    v16bf b = loadFragB(wrow, k0, h);
    acc = __builtin_amdgcn_wmma_f32_16x16x32_bf16(false, a, false, b, (short)0, acc, false, false);
  }
  float bn = bias[n0 + ln];
#pragma unroll
  for (int v = 0; v < 8; ++v) {
    int row = m0 + v + 8 * h;
    Out[(size_t)row * N + n0 + ln] = f2bf(acc[v] + bn);
  }
}

// ---------- pair-batched logits: out[p](M,N) = Abatch(M,K) @ Bbatch(N,K)^T ----------
// p = blockIdx.x ; i = p % nB ; j = p / nB ; sel? 0 -> use i, 1 -> use j
__global__ void pair_logits_wmma_kernel(const unsigned short* __restrict__ Abase,
                                        const unsigned short* __restrict__ Bbase,
                                        float* __restrict__ out,
                                        int M, int N, int K,
                                        int strideA, int strideB,
                                        int selA, int selB, int nB, int Nt) {
  int p = blockIdx.x;
  int i = p % nB, j = p / nB;
  const unsigned short* Ab = Abase + (size_t)(selA ? j : i) * strideA;
  const unsigned short* Bb = Bbase + (size_t)(selB ? j : i) * strideB;
  int lane = threadIdx.x & 31;
  int wv = threadIdx.x >> 5;
  int mt = wv / Nt, nt = wv % Nt;
  int m0 = mt * 16, n0 = nt * 16;
  int h = lane >> 4, ln = lane & 15;
  int ar = m0 + ln; if (ar > M - 1) ar = M - 1;
  int br = n0 + ln; if (br > N - 1) br = N - 1;
  const unsigned short* arow = Ab + (size_t)ar * K;
  const unsigned short* brow = Bb + (size_t)br * K;
  v8f acc = {0.f, 0.f, 0.f, 0.f, 0.f, 0.f, 0.f, 0.f};
  for (int k0 = 0; k0 < K; k0 += 32) {
    v16bf a = loadFragA(arow, k0, h);
    v16bf b = loadFragB(brow, k0, h);
    acc = __builtin_amdgcn_wmma_f32_16x16x32_bf16(false, a, false, b, (short)0, acc, false, false);
  }
  float* op = out + (size_t)p * M * N;
#pragma unroll
  for (int v = 0; v < 8; ++v) {
    int row = m0 + v + 8 * h;
    int col = n0 + ln;
    if (row < M && col < N) op[(size_t)row * N + col] = acc[v];
  }
}

// ---------- softmax(4*x) over last dim, in place ----------
__global__ void softmax4_kernel(float* __restrict__ x, int rows, int n) {
  int r = blockIdx.x * blockDim.x + threadIdx.x;
  if (r >= rows) return;
  float* p = x + (size_t)r * n;
  float mx = -1e30f;
  for (int i = 0; i < n; ++i) mx = fmaxf(mx, 4.0f * p[i]);
  float s = 0.f;
  for (int i = 0; i < n; ++i) { float e = __expf(4.0f * p[i] - mx); p[i] = e; s += e; }
  float inv = 1.0f / s;
  for (int i = 0; i < n; ++i) p[i] *= inv;
}

// ---------- fused trans1 + mblock -> sim_i2t[c][b][l] ----------
__global__ void sim_i2t_kernel(const float* __restrict__ img, const float* __restrict__ cap,
                               const float* __restrict__ a_i2t,
                               const float* __restrict__ o1w, const float* __restrict__ o1b,
                               const float* __restrict__ o2w, const float* __restrict__ o2b,
                               float* __restrict__ sim) {
  int idx = blockIdx.x;
  int l = idx % LL;
  int b = (idx / LL) % BB;
  int c = idx / (LL * BB);
  int t = threadIdx.x;
  __shared__ float wsh[RR];
  __shared__ float s_qv[16], s_qq[16], s_vv[16];
  __shared__ float cosb[16];
  if (t < RR) wsh[t] = a_i2t[((size_t)(c * BB + b) * RR + t) * LL + l];
  if (t < 16) { s_qv[t] = 0.f; s_qq[t] = 0.f; s_vv[t] = 0.f; }
  __syncthreads();
  const float* qrow = cap + ((size_t)c * LL + l) * DD;
  const float* itile = img + (size_t)b * RR * DD;
#pragma unroll
  for (int u = 0; u < 4; ++u) {
    int d = t + 256 * u;
    float v = 0.f;
    for (int r2 = 0; r2 < RR; ++r2) v += wsh[r2] * itile[r2 * DD + d];
    float q = qrow[d];
    int bb = d >> 6;
    atomicAdd(&s_qv[bb], q * v);
    atomicAdd(&s_qq[bb], q * q);
    atomicAdd(&s_vv[bb], v * v);
  }
  __syncthreads();
  if (t < 16) cosb[t] = s_qv[t] / fmaxf(sqrtf(s_qq[t]) * sqrtf(s_vv[t]), 1e-8f);
  __syncthreads();
  if (t == 0) {
    float o = o2b[0];
    for (int jj = 0; jj < 16; ++jj) {
      float hs = o1b[jj];
      for (int k = 0; k < 16; ++k) hs += o1w[jj * 16 + k] * cosb[k];
      o += o2w[jj] * tanhf(hs);
    }
    sim[idx] = o;
  }
}

// ---------- fused trans2 + mblock -> sim_t2i[c][b][r] ----------
__global__ void sim_t2i_kernel(const float* __restrict__ img, const float* __restrict__ cap,
                               const float* __restrict__ a_t2i,
                               const float* __restrict__ o1w, const float* __restrict__ o1b,
                               const float* __restrict__ o2w, const float* __restrict__ o2b,
                               float* __restrict__ sim) {
  int idx = blockIdx.x;
  int r = idx % RR;
  int b = (idx / RR) % BB;
  int c = idx / (RR * BB);
  int t = threadIdx.x;
  __shared__ float wsh[LL];
  __shared__ float s_qv[16], s_qq[16], s_vv[16];
  __shared__ float cosb[16];
  if (t < LL) wsh[t] = a_t2i[((size_t)(c * BB + b) * LL + t) * RR + r];
  if (t < 16) { s_qv[t] = 0.f; s_qq[t] = 0.f; s_vv[t] = 0.f; }
  __syncthreads();
  const float* qrow = img + ((size_t)b * RR + r) * DD;
  const float* ctile = cap + (size_t)c * LL * DD;
#pragma unroll
  for (int u = 0; u < 4; ++u) {
    int d = t + 256 * u;
    float v = 0.f;
    for (int l = 0; l < LL; ++l) v += wsh[l] * ctile[l * DD + d];
    float q = qrow[d];
    int bb = d >> 6;
    atomicAdd(&s_qv[bb], q * v);
    atomicAdd(&s_qq[bb], q * q);
    atomicAdd(&s_vv[bb], v * v);
  }
  __syncthreads();
  if (t < 16) cosb[t] = s_qv[t] / fmaxf(sqrtf(s_qq[t]) * sqrtf(s_vv[t]), 1e-8f);
  __syncthreads();
  if (t == 0) {
    float o = o2b[0];
    for (int jj = 0; jj < 16; ++jj) {
      float hs = o1b[jj];
      for (int k = 0; k < 16; ++k) hs += o1w[jj * 16 + k] * cosb[k];
      o += o2w[jj] * tanhf(hs);
    }
    sim[idx] = o;
  }
}

// ---------- re_txt = (a_i2t^T @ attn1) @ a_i2t vs a_txt1 -> d1f, d1c ----------
__global__ void re_txt_kernel(const float* __restrict__ a_i2t, const float* __restrict__ attn1,
                              const float* __restrict__ a_txt1,
                              float* __restrict__ d1f, float* __restrict__ d1c) {
  int p = blockIdx.x;
  int b = p % BB;
  int c = p / BB;
  int t = threadIdx.x;
  __shared__ float T[LL * RR];
  __shared__ float rowsum[LL];
  __shared__ float tot;
  const float* A = a_i2t + (size_t)p * RR * LL;        // [r][l]
  const float* At = attn1 + (size_t)b * RR * RR;       // [r][s]
  for (int e = t; e < LL * RR; e += 256) {
    int l = e / RR, s2 = e % RR;
    float acc = 0.f;
    for (int r2 = 0; r2 < RR; ++r2) acc += A[r2 * LL + l] * At[r2 * RR + s2];
    T[e] = acc;
  }
  if (t < LL) rowsum[t] = 0.f;
  if (t == 0) tot = 0.f;
  __syncthreads();
  const float* ax = a_txt1 + (size_t)c * LL * LL;
  float mytot = 0.f;
  for (int e = t; e < LL * LL; e += 256) {
    int l = e / LL, m = e % LL;
    float acc = 0.f;
    for (int s2 = 0; s2 < RR; ++s2) acc += T[l * RR + s2] * A[s2 * LL + m];
    float dd = acc - ax[e];
    float d2 = dd * dd;
    atomicAdd(&rowsum[l], d2);
    mytot += d2;
  }
  atomicAdd(&tot, mytot);
  __syncthreads();
  if (t < LL) d1f[(size_t)p * LL + t] = sqrtf(rowsum[t]);
  if (t == 0) d1c[p] = sqrtf(tot);
}

// ---------- re_img = (a_t2i^T(l->) @ a_txt2) @ a_t2i vs attn2 -> d2f, d2c ----------
__global__ void re_img_kernel(const float* __restrict__ a_t2i, const float* __restrict__ attn2,
                              const float* __restrict__ a_txt2,
                              float* __restrict__ d2f, float* __restrict__ d2c) {
  int p = blockIdx.x;
  int b = p % BB;
  int c = p / BB;
  int t = threadIdx.x;
  __shared__ float U[RR * LL];   // [r][k]
  __shared__ float rowsum[RR];
  __shared__ float tot;
  const float* A = a_t2i + (size_t)p * LL * RR;        // [l][r]
  const float* ax2 = a_txt2 + (size_t)c * LL * LL;     // [l][k]
  for (int e = t; e < RR * LL; e += 256) {
    int r2 = e / LL, k = e % LL;
    float acc = 0.f;
    for (int l = 0; l < LL; ++l) acc += A[l * RR + r2] * ax2[l * LL + k];
    U[e] = acc;
  }
  if (t < RR) rowsum[t] = 0.f;
  if (t == 0) tot = 0.f;
  __syncthreads();
  const float* At2 = attn2 + (size_t)b * RR * RR;
  float mytot = 0.f;
  for (int e = t; e < RR * RR; e += 256) {
    int r2 = e / RR, s2 = e % RR;
    float acc = 0.f;
    for (int k = 0; k < LL; ++k) acc += U[r2 * LL + k] * A[k * RR + s2];
    float dd = acc - At2[e];
    float d2v = dd * dd;
    atomicAdd(&rowsum[r2], d2v);
    mytot += d2v;
  }
  atomicAdd(&tot, mytot);
  __syncthreads();
  if (t < RR) d2f[(size_t)p * RR + t] = sqrtf(rowsum[t]);
  if (t == 0) d2c[p] = sqrtf(tot);
}

// ---------- final reduction: s.T, si.T, st.T concatenated ----------
__global__ void final_kernel(const float* __restrict__ sim_i2t, const float* __restrict__ sim_t2i,
                             const float* __restrict__ d1f, const float* __restrict__ d1c,
                             const float* __restrict__ d2f, const float* __restrict__ d2c,
                             const float* __restrict__ w_i2t, const float* __restrict__ w_t2i,
                             float* __restrict__ out) {
  int p = blockIdx.x * blockDim.x + threadIdx.x;
  if (p >= NC * BB) return;
  int b = p % BB, c = p / BB;
  float mi = 0.f, sa = 0.f;
  for (int l = 0; l < LL; ++l) {
    float sv = sim_i2t[(size_t)p * LL + l];
    float wi = w_i2t[l % 12];
    sa += sv * wi + 0.05f * fmaxf(0.1f - d1f[(size_t)p * LL + l], 0.f);
    mi += sv;
  }
  sa *= (1.f / LL); mi *= (1.f / LL);
  float mt_ = 0.f, sb = 0.f;
  for (int r2 = 0; r2 < RR; ++r2) {
    float sv = sim_t2i[(size_t)p * RR + r2];
    sb += sv * w_t2i[r2] + 0.05f * fmaxf(0.1f - d2f[(size_t)p * RR + r2], 0.f);
    mt_ += sv;
  }
  sb *= (1.f / RR); mt_ *= (1.f / RR);
  out[(size_t)b * NC + c] = sa + sb;
  out[(size_t)(NC * BB) + b * NC + c] = mi + 0.01f * fmaxf(0.5f - d1c[p], 0.f);
  out[(size_t)(2 * NC * BB) + b * NC + c] = mt_ + 0.01f * fmaxf(0.5f - d2c[p], 0.f);
}

extern "C" void kernel_launch(void* const* d_in, const int* in_sizes, int n_in,
                              void* d_out, int out_size, void* d_ws, size_t ws_size,
                              hipStream_t stream) {
  (void)in_sizes; (void)n_in; (void)out_size; (void)ws_size;
  const float* img = (const float*)d_in[0];
  const float* cap = (const float*)d_in[1];
  const float* i1w = (const float*)d_in[3];
  const float* i1b = (const float*)d_in[4];
  const float* i2w = (const float*)d_in[5];
  const float* i2b = (const float*)d_in[6];
  const float* t1w = (const float*)d_in[7];
  const float* t1b = (const float*)d_in[8];
  const float* t2w = (const float*)d_in[9];
  const float* t2b = (const float*)d_in[10];
  const float* o1w = (const float*)d_in[11];
  const float* o1b = (const float*)d_in[12];
  const float* o2w = (const float*)d_in[13];
  const float* o2b = (const float*)d_in[14];
  const float* wi2t = (const float*)d_in[15];
  const float* wt2i = (const float*)d_in[16];
  float* out = (float*)d_out;

  const int nImg = BB * RR;        // 1728 rows
  const int nCap = NC * LL;        // 1152 rows
  const int imgStride = RR * DD;   // elements per image batch
  const int capStride = LL * DD;   // elements per caption

  char* base = (char*)d_ws;
  size_t off = 0;
  auto alloc = [&](size_t bytes) -> char* {
    char* p = base + off;
    off += (bytes + 255) & ~(size_t)255;
    return p;
  };
  unsigned short* img_bf = (unsigned short*)alloc((size_t)nImg * DD * 2);
  unsigned short* cap_bf = (unsigned short*)alloc((size_t)nCap * DD * 2);
  unsigned short* i1w_bf = (unsigned short*)alloc((size_t)DD * DD * 2);
  unsigned short* i2w_bf = (unsigned short*)alloc((size_t)DD * DD * 2);
  unsigned short* t1w_bf = (unsigned short*)alloc((size_t)DD * DD * 2);
  unsigned short* t2w_bf = (unsigned short*)alloc((size_t)DD * DD * 2);
  unsigned short* v_img1 = (unsigned short*)alloc((size_t)nImg * DD * 2);
  unsigned short* v_img2 = (unsigned short*)alloc((size_t)nImg * DD * 2);
  unsigned short* v_t1   = (unsigned short*)alloc((size_t)nCap * DD * 2);
  unsigned short* v_t2   = (unsigned short*)alloc((size_t)nCap * DD * 2);
  float* attn1  = (float*)alloc((size_t)BB * RR * RR * 4);
  float* attn2  = (float*)alloc((size_t)BB * RR * RR * 4);
  float* a_i2t  = (float*)alloc((size_t)NC * BB * RR * LL * 4);
  float* a_t2i  = (float*)alloc((size_t)NC * BB * LL * RR * 4);
  float* a_txt1 = (float*)alloc((size_t)NC * LL * LL * 4);
  float* a_txt2 = (float*)alloc((size_t)NC * LL * LL * 4);
  float* sim_i2t = (float*)alloc((size_t)NC * BB * LL * 4);
  float* sim_t2i = (float*)alloc((size_t)NC * BB * RR * 4);
  float* d1f = (float*)alloc((size_t)NC * BB * LL * 4);
  float* d1c = (float*)alloc((size_t)NC * BB * 4);
  float* d2f = (float*)alloc((size_t)NC * BB * RR * 4);
  float* d2c = (float*)alloc((size_t)NC * BB * 4);

  // 1) casts to bf16
  cast_bf16_kernel<<<1024, 256, 0, stream>>>(img, img_bf, nImg * DD);
  cast_bf16_kernel<<<1024, 256, 0, stream>>>(cap, cap_bf, nCap * DD);
  cast_bf16_kernel<<<1024, 256, 0, stream>>>(i1w, i1w_bf, DD * DD);
  cast_bf16_kernel<<<1024, 256, 0, stream>>>(i2w, i2w_bf, DD * DD);
  cast_bf16_kernel<<<1024, 256, 0, stream>>>(t1w, t1w_bf, DD * DD);
  cast_bf16_kernel<<<1024, 256, 0, stream>>>(t2w, t2w_bf, DD * DD);

  // 2) dense linears (WMMA)
  linear_wmma_kernel<<<dim3(nImg / 32, DD / 64), 256, 0, stream>>>(img_bf, i1w_bf, i1b, v_img1, nImg, DD, DD);
  linear_wmma_kernel<<<dim3(nImg / 32, DD / 64), 256, 0, stream>>>(img_bf, i2w_bf, i2b, v_img2, nImg, DD, DD);
  linear_wmma_kernel<<<dim3(nCap / 32, DD / 64), 256, 0, stream>>>(cap_bf, t1w_bf, t1b, v_t1, nCap, DD, DD);
  linear_wmma_kernel<<<dim3(nCap / 32, DD / 64), 256, 0, stream>>>(cap_bf, t2w_bf, t2b, v_t2, nCap, DD, DD);

  // 3) batched similarity logits (WMMA) + softmax(4x)
  // attn_img1 / attn_img2: per b, img_b @ v_img*_b^T  (36x36)
  pair_logits_wmma_kernel<<<BB, 32 * 3 * 3, 0, stream>>>(img_bf, v_img1, attn1, RR, RR, DD,
                                                         imgStride, imgStride, 0, 0, BB, 3);
  pair_logits_wmma_kernel<<<BB, 32 * 3 * 3, 0, stream>>>(img_bf, v_img2, attn2, RR, RR, DD,
                                                         imgStride, imgStride, 0, 0, BB, 3);
  softmax4_kernel<<<CDIV(BB * RR, 256), 256, 0, stream>>>(attn1, BB * RR, RR);
  softmax4_kernel<<<CDIV(BB * RR, 256), 256, 0, stream>>>(attn2, BB * RR, RR);
  // a_txt1 / a_txt2: per c, cap_c @ v_t*_c^T (24x24)
  pair_logits_wmma_kernel<<<NC, 32 * 2 * 2, 0, stream>>>(cap_bf, v_t1, a_txt1, LL, LL, DD,
                                                         capStride, capStride, 0, 0, NC, 2);
  pair_logits_wmma_kernel<<<NC, 32 * 2 * 2, 0, stream>>>(cap_bf, v_t2, a_txt2, LL, LL, DD,
                                                         capStride, capStride, 0, 0, NC, 2);
  softmax4_kernel<<<CDIV(NC * LL, 256), 256, 0, stream>>>(a_txt1, NC * LL, LL);
  softmax4_kernel<<<CDIV(NC * LL, 256), 256, 0, stream>>>(a_txt2, NC * LL, LL);
  // a_i2t: p = c*48+b -> img_b @ v_t1_c^T (36x24)
  pair_logits_wmma_kernel<<<NC * BB, 32 * 3 * 2, 0, stream>>>(img_bf, v_t1, a_i2t, RR, LL, DD,
                                                              imgStride, capStride, 0, 1, BB, 2);
  softmax4_kernel<<<CDIV(NC * BB * RR, 256), 256, 0, stream>>>(a_i2t, NC * BB * RR, LL);
  // a_t2i: p = c*48+b -> cap_c @ v_img2_b^T (24x36)
  pair_logits_wmma_kernel<<<NC * BB, 32 * 2 * 3, 0, stream>>>(cap_bf, v_img2, a_t2i, LL, RR, DD,
                                                              capStride, imgStride, 1, 0, BB, 3);
  softmax4_kernel<<<CDIV(NC * BB * LL, 256), 256, 0, stream>>>(a_t2i, NC * BB * LL, RR);

  // 4) fused trans + mblock similarities
  sim_i2t_kernel<<<NC * BB * LL, 256, 0, stream>>>(img, cap, a_i2t, o1w, o1b, o2w, o2b, sim_i2t);
  sim_t2i_kernel<<<NC * BB * RR, 256, 0, stream>>>(img, cap, a_t2i, o1w, o1b, o2w, o2b, sim_t2i);

  // 5) reconstruction distances
  re_txt_kernel<<<NC * BB, 256, 0, stream>>>(a_i2t, attn1, a_txt1, d1f, d1c);
  re_img_kernel<<<NC * BB, 256, 0, stream>>>(a_t2i, attn2, a_txt2, d2f, d2c);

  // 6) final scores
  final_kernel<<<CDIV(NC * BB, 256), 256, 0, stream>>>(sim_i2t, sim_t2i, d1f, d1c, d2f, d2c,
                                                       wi2t, wt2i, out);
}
